// GraphWaveNet_86199993631187
// MI455X (gfx1250) — matrix operand order repaired and checked by reading, not compiled, not run on hardware
//
#include <hip/hip_runtime.h>
#include <hip/hip_bf16.h>
#include <math.h>

// ---------------- problem constants ----------------
#define NB 8
#define NT 64
#define NNODE 370
#define NH 64
#define NL 8
#define NP 384          // padded node dim (multiple of 32)
#define BT (NB*NT)      // 512
#define BN (NB*NNODE)   // 2960
#define MROWS (BT*NNODE) // 189440
#define KEEG (NNODE*NH)  // 23680
#define KTOPC 37

typedef _Float16 v16h __attribute__((ext_vector_type(16)));
typedef float    v8f  __attribute__((ext_vector_type(8)));

// ---------------- WMMA helpers ----------------
__device__ __forceinline__ v8f wmma32(v16h a, v16h b, v8f c) {
  return __builtin_amdgcn_wmma_f32_16x16x32_f16(false, a, false, b, (short)0, c,
                                                false, false);
}

__device__ __forceinline__ float gelu_f(float x) {
  return 0.5f * x * (1.0f + erff(x * 0.70710678118654752f));
}

// hardware transcendental paths on gfx1250 (V_TANH_F32 / V_EXP_F32 / V_RCP_F32),
// guarded so missing builtins fall back to libm instead of breaking the build.
__device__ __forceinline__ float fast_tanh(float x) {
#if __has_builtin(__builtin_amdgcn_tanhf)
  return __builtin_amdgcn_tanhf(x);
#else
  return tanhf(x);
#endif
}
__device__ __forceinline__ float fast_sigmoid(float x) {
#if __has_builtin(__builtin_amdgcn_rcpf) && __has_builtin(__builtin_amdgcn_exp2f)
  return __builtin_amdgcn_rcpf(1.0f + __builtin_amdgcn_exp2f(-1.44269504088896f * x));
#else
  return 1.0f / (1.0f + expf(-x));
#endif
}

// A fragment 16x32 f16 from f32 row-major. rowptr = row for this lane.
//   i in [0,8):  k = k0 + i + 8*hi
//   i in [8,16): k = k0 + 16 + (i-8) + 8*hi
__device__ __forceinline__ v16h a_frag_f32(const float* __restrict__ rowptr,
                                           int k0, int hi, bool valid, int doGelu) {
  v16h a;
  if (valid) {
    const float* p = rowptr + k0 + 8 * hi;
#pragma unroll
    for (int i = 0; i < 8; ++i) {
      float v0 = p[i];
      float v1 = p[16 + i];
      if (doGelu) { v0 = gelu_f(v0); v1 = gelu_f(v1); }
      a[i]     = (_Float16)v0;
      a[8 + i] = (_Float16)v1;
    }
  } else {
#pragma unroll
    for (int i = 0; i < 16; ++i) a[i] = (_Float16)0.0f;
  }
  return a;
}

__device__ __forceinline__ v16h a_frag_f16(const _Float16* __restrict__ rowptr,
                                           int k0, int hi, bool valid) {
  v16h a;
  if (valid) {
    const _Float16* p = rowptr + k0 + 8 * hi;
#pragma unroll
    for (int i = 0; i < 8; ++i) {
      a[i]     = p[i];
      a[8 + i] = p[16 + i];
    }
  } else {
#pragma unroll
    for (int i = 0; i < 16; ++i) a[i] = (_Float16)0.0f;
  }
  return a;
}

// B fragment 32x16 f16 from "Bt" layout: Bt[o*pitchK + k] (out-major, K contiguous).
// lane: col = n0 + (lane&15); element i<8: k = k0 + i + 8*hi; i>=8: k = k0 + 16 + (i-8) + 8*hi.
__device__ __forceinline__ v16h b_frag(const _Float16* __restrict__ Bt,
                                       int pitchK, int n0, int k0, int lane) {
  int col = n0 + (lane & 15);
  int hi  = lane >> 4;
  const _Float16* p = Bt + (size_t)col * pitchK + k0 + 8 * hi;
  v16h b;
#pragma unroll
  for (int i = 0; i < 8; ++i) { b[i] = p[i]; b[8 + i] = p[16 + i]; }
  return b;
}

// ---------------- small utility kernels ----------------
__global__ void k_fill_f32(float* p, size_t n) {
  size_t i = (size_t)blockIdx.x * 256 + threadIdx.x;
  if (i < n) p[i] = 0.0f;
}

// dst[o*I+i] = (f16) src[i*O+o]   (transpose-convert)
__global__ void k_cvt_T(_Float16* dst, const float* src, int I, int O) {
  long idx = (long)blockIdx.x * 256 + threadIdx.x;
  long n = (long)I * O;
  if (idx >= n) return;
  long o = idx / I, i = idx % I;
  dst[idx] = (_Float16)src[i * (long)O + o];
}

// dst[j] = (f16) src[j*stride + offset]   (strided gather convert; stride=1 -> plain)
__global__ void k_cvt_S(_Float16* dst, const float* src, long n, int stride, int offset) {
  long idx = (long)blockIdx.x * 256 + threadIdx.x;
  if (idx >= n) return;
  dst[idx] = (_Float16)src[idx * (long)stride + offset];
}

// scatter edges: A[dst*N + src] += w
__global__ void k_scatter(const int* ei, const float* ew, int E, float* ADJ) {
  int e = blockIdx.x * 256 + threadIdx.x;
  if (e >= E) return;
  int s = ei[e];
  int d = ei[E + e];
  atomicAdd(&ADJ[(size_t)d * NNODE + s], ew[e]);
}

// per-row: self-loop fix + D^-1/2. one wave per row.
__global__ void k_rowdeg(float* ADJ, float* DINV) {
  int r = blockIdx.x;
  int lane = threadIdx.x;
  float s = 0.0f;
  for (int c = lane; c < NNODE; c += 32) s += ADJ[(size_t)r * NNODE + c];
#pragma unroll
  for (int o = 16; o; o >>= 1) s += __shfl_xor(s, o, 32);
  if (lane == 0) {
    float d0 = ADJ[(size_t)r * NNODE + r];
    if (!(d0 > 0.0f)) { ADJ[(size_t)r * NNODE + r] = d0 + 1.0f; s += 1.0f; }
    DINV[r] = (s > 0.0f) ? rsqrtf(s) : 0.0f;
  }
}

// out_f16[r*NP+c] = (r<N && c<N) ? ADJ[r,c]*dinv[r]*dinv[c] : 0
__global__ void k_scale_cvt(const float* ADJ, const float* DINV, _Float16* OUT) {
  int idx = blockIdx.x * 256 + threadIdx.x;
  if (idx >= NP * NP) return;
  int r = idx / NP, c = idx % NP;
  float v = 0.0f;
  if (r < NNODE && c < NNODE) v = ADJ[(size_t)r * NNODE + c] * DINV[r] * DINV[c];
  OUT[idx] = (_Float16)v;
}

// adaptive adjacency: softmax(relu(S T^T)) row-wise + top-k mask. one block/row.
__global__ void k_adapt(const float* __restrict__ es, const float* __restrict__ et,
                        float* __restrict__ ADJ) {
  __shared__ float vals[NP];
  __shared__ float red[16];
  int r = blockIdx.x;
  int c = threadIdx.x;          // 0..383
  int lane = c & 31, wv = c >> 5;
  float sc = -3.0e38f;
  if (c < NNODE) {
    const float* a = es + (size_t)r * 16;
    const float* b = et + (size_t)c * 16;
    float acc = 0.0f;
#pragma unroll
    for (int i = 0; i < 16; ++i) acc = fmaf(a[i], b[i], acc);
    sc = fmaxf(acc, 0.0f);      // relu
  }
  // block max
  float m = sc;
#pragma unroll
  for (int o = 16; o; o >>= 1) m = fmaxf(m, __shfl_xor(m, o, 32));
  if (lane == 0) red[wv] = m;
  __syncthreads();
  if (c == 0) { float mm = red[0]; for (int w = 1; w < 12; ++w) mm = fmaxf(mm, red[w]); red[0] = mm; }
  __syncthreads();
  float mx = red[0];
  float e = (c < NNODE) ? expf(sc - mx) : 0.0f;
  __syncthreads();
  float s = e;
#pragma unroll
  for (int o = 16; o; o >>= 1) s += __shfl_xor(s, o, 32);
  if (lane == 0) red[wv] = s;
  __syncthreads();
  if (c == 0) { float ss = 0.0f; for (int w = 0; w < 12; ++w) ss += red[w]; red[0] = ss; }
  __syncthreads();
  float p = e / red[0];
  vals[c] = p;
  __syncthreads();
  if (c < NNODE) {
    int cnt = 0;
    for (int j = 0; j < NNODE; ++j) {
      float vj = vals[j];
      cnt += (vj > p) || (vj == p && j < c);
    }
    ADJ[(size_t)r * NNODE + c] = (cnt < KTOPC) ? p : 0.0f;
  }
}

// ---------------- WMMA GEMM kernels (128 threads = 4 waves, 1 tile/wave) -------
// Generic: C(Mx64) = act(A f32, Mx64) @ Bt(64x64 f16, Bt[o][k]) + bias
__global__ void k_gemm64(const float* __restrict__ A, const _Float16* __restrict__ Bt,
                         const float* __restrict__ bias, float* Cf32, _Float16* Cf16,
                         long M, int aGelu) {
  int lane = threadIdx.x & 31;
  int wave = threadIdx.x >> 5;
  int hi = lane >> 4, colL = lane & 15;
  long m0 = ((long)blockIdx.x * 4 + wave) * 16;
  long row = m0 + (lane & 15);
  bool valid = row < M;
  const float* rp = A + row * 64;
  v8f acc[4];
#pragma unroll
  for (int n = 0; n < 4; ++n)
#pragma unroll
    for (int v = 0; v < 8; ++v) acc[n][v] = 0.0f;
#pragma unroll
  for (int k0 = 0; k0 < 64; k0 += 32) {
    v16h a = a_frag_f32(rp, k0, hi, valid, aGelu);
#pragma unroll
    for (int n = 0; n < 4; ++n)
      acc[n] = wmma32(a, b_frag(Bt, 64, n * 16, k0, lane), acc[n]);
  }
#pragma unroll
  for (int n = 0; n < 4; ++n) {
    int col = n * 16 + colL;
    float bv = bias ? bias[col] : 0.0f;
#pragma unroll
    for (int v = 0; v < 8; ++v) {
      long r = m0 + 8 * hi + v;
      if (r < M) {
        float val = acc[n][v] + bv;
        if (Cf32) Cf32[r * 64 + col] = val;
        if (Cf16) Cf16[r * 64 + col] = (_Float16)val;
      }
    }
  }
}

// GCN pre-projection: per slice bt, P[bt][h][n_pad] = (X[bt] @ Wt^T) transposed.
__global__ void k_gcn_pre(const float* __restrict__ X, const _Float16* __restrict__ Wt,
                          _Float16* __restrict__ P) {
  int lane = threadIdx.x & 31;
  int wave = threadIdx.x >> 5;
  int hi = lane >> 4, colL = lane & 15;
  int bt = blockIdx.x;
  int m0 = (blockIdx.y * 4 + wave) * 16;
  const float* Ab = X + (size_t)bt * NNODE * 64;
  int rowm = m0 + (lane & 15);
  bool valid = rowm < NNODE;
  const float* rp = Ab + (size_t)rowm * 64;
  v8f acc[4];
#pragma unroll
  for (int n = 0; n < 4; ++n)
#pragma unroll
    for (int v = 0; v < 8; ++v) acc[n][v] = 0.0f;
#pragma unroll
  for (int k0 = 0; k0 < 64; k0 += 32) {
    v16h a = a_frag_f32(rp, k0, hi, valid, 0);
#pragma unroll
    for (int n = 0; n < 4; ++n)
      acc[n] = wmma32(a, b_frag(Wt, 64, n * 16, k0, lane), acc[n]);
  }
  _Float16* pb = P + (size_t)bt * 64 * NP;
#pragma unroll
  for (int n = 0; n < 4; ++n) {
    int col = n * 16 + colL;
    _Float16* q = pb + (size_t)col * NP + m0 + 8 * hi;
#pragma unroll
    for (int v = 0; v < 8; ++v) q[v] = (_Float16)acc[n][v]; // contiguous transposed store
  }
}

// h = AdjF @ PF + AdjA @ PA + (bf+ba); scatter-store f16 into CIN[(b,n,t,h)]
__global__ void k_adj_gemm(const _Float16* __restrict__ AdjF, const _Float16* __restrict__ AdjA,
                           const _Float16* __restrict__ PF, const _Float16* __restrict__ PA,
                           const float* __restrict__ bf, const float* __restrict__ ba,
                           _Float16* __restrict__ CIN) {
  int lane = threadIdx.x & 31;
  int wave = threadIdx.x >> 5;
  int hi = lane >> 4, colL = lane & 15;
  int bt = blockIdx.x;
  int m0 = (blockIdx.y * 4 + wave) * 16;
  int b = bt >> 6, t = bt & 63;
  const _Float16* pF = PF + (size_t)bt * 64 * NP;
  const _Float16* pA = PA + (size_t)bt * 64 * NP;
  int rowm = m0 + (lane & 15);
  const _Float16* arF = AdjF + (size_t)rowm * NP;
  const _Float16* arA = AdjA + (size_t)rowm * NP;
  v8f acc[4];
#pragma unroll
  for (int n = 0; n < 4; ++n)
#pragma unroll
    for (int v = 0; v < 8; ++v) acc[n][v] = 0.0f;
  for (int k0 = 0; k0 < NP; k0 += 32) {
    v16h aF = a_frag_f16(arF, k0, hi, true);
    v16h aA = a_frag_f16(arA, k0, hi, true);
#pragma unroll
    for (int n = 0; n < 4; ++n) {
      acc[n] = wmma32(aF, b_frag(pF, NP, n * 16, k0, lane), acc[n]);
      acc[n] = wmma32(aA, b_frag(pA, NP, n * 16, k0, lane), acc[n]);
    }
  }
#pragma unroll
  for (int n = 0; n < 4; ++n) {
    int col = n * 16 + colL;
    float bv = bf[col] + ba[col];
#pragma unroll
    for (int v = 0; v < 8; ++v) {
      int node = m0 + 8 * hi + v;
      if (node < NNODE) {
        size_t addr = (((size_t)(b * NNODE + node)) * 64 + t) * 64 + col;
        CIN[addr] = (_Float16)(acc[n][v] + bv);
      }
    }
  }
}

// gated causal conv (K=2): out = tanh(Xs@Wf0 + X@Wf1 + fb) * sigmoid(Xs@Wg0 + X@Wg1 + gb)
__global__ void k_conv_gemm(const _Float16* __restrict__ CIN,
                            const _Float16* Wf1, const _Float16* Wf0,
                            const _Float16* Wg1, const _Float16* Wg0,
                            const float* __restrict__ fb, const float* __restrict__ gb,
                            int dil, _Float16* __restrict__ GT) {
  int lane = threadIdx.x & 31;
  int wave = threadIdx.x >> 5;
  int hi = lane >> 4, colL = lane & 15;
  int bn = blockIdx.x;
  int t0 = wave * 16;
  const _Float16* base = CIN + (size_t)bn * 64 * 64;
  int t = t0 + (lane & 15);
  int ts = t - dil;
  v8f aF[4], aG[4];
#pragma unroll
  for (int n = 0; n < 4; ++n)
#pragma unroll
    for (int v = 0; v < 8; ++v) { aF[n][v] = 0.0f; aG[n][v] = 0.0f; }
#pragma unroll
  for (int k0 = 0; k0 < 64; k0 += 32) {
    v16h ac = a_frag_f16(base + (size_t)t * 64, k0, hi, true);
    v16h as = a_frag_f16(base + (size_t)(ts < 0 ? 0 : ts) * 64, k0, hi, ts >= 0);
#pragma unroll
    for (int n = 0; n < 4; ++n) {
      aF[n] = wmma32(ac, b_frag(Wf1, 64, n * 16, k0, lane), aF[n]);
      aF[n] = wmma32(as, b_frag(Wf0, 64, n * 16, k0, lane), aF[n]);
      aG[n] = wmma32(ac, b_frag(Wg1, 64, n * 16, k0, lane), aG[n]);
      aG[n] = wmma32(as, b_frag(Wg0, 64, n * 16, k0, lane), aG[n]);
    }
  }
#pragma unroll
  for (int n = 0; n < 4; ++n) {
    int col = n * 16 + colL;
    float fbv = fb[col], gbv = gb[col];
#pragma unroll
    for (int v = 0; v < 8; ++v) {
      int tr = t0 + 8 * hi + v;
      float val = fast_tanh(aF[n][v] + fbv) * fast_sigmoid(aG[n][v] + gbv);
      GT[((size_t)bn * 64 + tr) * 64 + col] = (_Float16)val;
    }
  }
}

// skip/res 1x1 convs; scatter into (b,t,n,h). SKIP accumulated, RES stored.
__global__ void k_skipres(const _Float16* __restrict__ GT,
                          const _Float16* St, const _Float16* Rt,
                          const float* __restrict__ sb, const float* __restrict__ rb,
                          float* __restrict__ SKIP, float* __restrict__ RES) {
  int lane = threadIdx.x & 31;
  int wave = threadIdx.x >> 5;
  int hi = lane >> 4, colL = lane & 15;
  int bn = blockIdx.x;
  int t0 = wave * 16;
  int b = bn / NNODE, node = bn % NNODE;
  const _Float16* base = GT + (size_t)bn * 64 * 64;
  int t = t0 + (lane & 15);
  v8f aS[4], aR[4];
#pragma unroll
  for (int n = 0; n < 4; ++n)
#pragma unroll
    for (int v = 0; v < 8; ++v) { aS[n][v] = 0.0f; aR[n][v] = 0.0f; }
#pragma unroll
  for (int k0 = 0; k0 < 64; k0 += 32) {
    v16h a = a_frag_f16(base + (size_t)t * 64, k0, hi, true);
#pragma unroll
    for (int n = 0; n < 4; ++n) {
      aS[n] = wmma32(a, b_frag(St, 64, n * 16, k0, lane), aS[n]);
      aR[n] = wmma32(a, b_frag(Rt, 64, n * 16, k0, lane), aR[n]);
    }
  }
#pragma unroll
  for (int n = 0; n < 4; ++n) {
    int col = n * 16 + colL;
    float sbv = sb[col], rbv = rb[col];
#pragma unroll
    for (int v = 0; v < 8; ++v) {
      int tr = t0 + 8 * hi + v;
      size_t addr = (((size_t)(b * 64 + tr)) * NNODE + node) * 64 + col;
      SKIP[addr] += aS[n][v] + sbv;
      RES[addr]  = aR[n][v] + rbv;
    }
  }
}

// X = LayerNorm(RES + X) * g + b   over H=64; one wave per row.
__global__ void k_layernorm(float* __restrict__ X, const float* __restrict__ RES,
                            const float* __restrict__ g, const float* __restrict__ b,
                            long M) {
  int lane = threadIdx.x & 31;
  int wv = threadIdx.x >> 5;
  long row = (long)blockIdx.x * 8 + wv;
  if (row >= M) return;
  float* r1 = X + row * 64;
  const float* r2 = RES + row * 64;
  float x0 = r1[lane] + r2[lane];
  float x1 = r1[lane + 32] + r2[lane + 32];
  float s = x0 + x1, ss = x0 * x0 + x1 * x1;
#pragma unroll
  for (int o = 16; o; o >>= 1) { s += __shfl_xor(s, o, 32); ss += __shfl_xor(ss, o, 32); }
  float mean = s * (1.0f / 64.0f);
  float var = ss * (1.0f / 64.0f) - mean * mean;
  float inv = rsqrtf(var + 1e-5f);
  r1[lane]      = (x0 - mean) * inv * g[lane] + b[lane];
  r1[lane + 32] = (x1 - mean) * inv * g[lane + 32] + b[lane + 32];
}

// EEG head stage 1: (512 x 23680) @ Bt(256 x 23680) + b, GELU, f16 out (512x256)
// 128 threads: wave -> 64-col quadrant.
__global__ void k_eeg1(const _Float16* __restrict__ A, const _Float16* __restrict__ Bt,
                       const float* __restrict__ bias, _Float16* __restrict__ C) {
  int lane = threadIdx.x & 31;
  int wave = threadIdx.x >> 5;
  int hi = lane >> 4, colL = lane & 15;
  long m0 = (long)blockIdx.x * 16;
  int n0 = wave * 64;
  const _Float16* rp = A + (size_t)(m0 + (lane & 15)) * KEEG;
  v8f acc[4];
#pragma unroll
  for (int n = 0; n < 4; ++n)
#pragma unroll
    for (int v = 0; v < 8; ++v) acc[n][v] = 0.0f;
  for (int k0 = 0; k0 < KEEG; k0 += 32) {
    __builtin_prefetch(rp + k0 + 512, 0, 1);
    v16h a = a_frag_f16(rp, k0, hi, true);
#pragma unroll
    for (int n = 0; n < 4; ++n)
      acc[n] = wmma32(a, b_frag(Bt, KEEG, n0 + n * 16, k0, lane), acc[n]);
  }
#pragma unroll
  for (int n = 0; n < 4; ++n) {
    int col = n0 + n * 16 + colL;
    float bv = bias[col];
#pragma unroll
    for (int v = 0; v < 8; ++v) {
      long r = m0 + 8 * hi + v;
      C[r * 256 + col] = (_Float16)gelu_f(acc[n][v] + bv);
    }
  }
}

// EEG head stage 2: (512 x 256) @ Bt(64 x 256) + b -> f32 out (512 x 64)
__global__ void k_eeg2(const _Float16* __restrict__ A, const _Float16* __restrict__ Bt,
                       const float* __restrict__ bias, float* __restrict__ OUT) {
  int lane = threadIdx.x & 31;
  int wave = threadIdx.x >> 5;
  int hi = lane >> 4, colL = lane & 15;
  long m0 = ((long)blockIdx.x * 4 + wave) * 16;
  const _Float16* rp = A + (size_t)(m0 + (lane & 15)) * 256;
  v8f acc[4];
#pragma unroll
  for (int n = 0; n < 4; ++n)
#pragma unroll
    for (int v = 0; v < 8; ++v) acc[n][v] = 0.0f;
#pragma unroll
  for (int k0 = 0; k0 < 256; k0 += 32) {
    v16h a = a_frag_f16(rp, k0, hi, true);
#pragma unroll
    for (int n = 0; n < 4; ++n)
      acc[n] = wmma32(a, b_frag(Bt, 256, n * 16, k0, lane), acc[n]);
  }
#pragma unroll
  for (int n = 0; n < 4; ++n) {
    int col = n * 16 + colL;
    float bv = bias[col];
#pragma unroll
    for (int v = 0; v < 8; ++v) {
      long r = m0 + 8 * hi + v;
      OUT[r * 64 + col] = acc[n][v] + bv;
    }
  }
}

// ---------------- host side ----------------
static inline size_t al256(size_t x) { return (x + 255) & ~(size_t)255; }

extern "C" void kernel_launch(void* const* d_in, const int* in_sizes, int n_in,
                              void* d_out, int out_size, void* d_ws, size_t ws_size,
                              hipStream_t stream) {
  const float* x        = (const float*)d_in[0];
  const int*   eidx     = (const int*)d_in[1];
  const float* ew       = (const float*)d_in[2];
  const float* in_W     = (const float*)d_in[3];
  const float* in_b     = (const float*)d_in[4];
  const float* gcnf_W   = (const float*)d_in[5];
  const float* gcnf_b   = (const float*)d_in[6];
  const float* emb_src  = (const float*)d_in[7];
  const float* emb_tgt  = (const float*)d_in[8];
  const float* gcna_W   = (const float*)d_in[9];
  const float* gcna_b   = (const float*)d_in[10];
  const float* filt_W   = (const float*)d_in[11];
  const float* filt_b   = (const float*)d_in[12];
  const float* gate_W   = (const float*)d_in[13];
  const float* gate_b   = (const float*)d_in[14];
  const float* res_W    = (const float*)d_in[15];
  const float* res_b    = (const float*)d_in[16];
  const float* skip_W   = (const float*)d_in[17];
  const float* skip_b   = (const float*)d_in[18];
  const float* ln_g     = (const float*)d_in[19];
  const float* ln_b     = (const float*)d_in[20];
  const float* end1_W   = (const float*)d_in[21];
  const float* end1_b   = (const float*)d_in[22];
  const float* end2_W   = (const float*)d_in[23];
  const float* end2_b   = (const float*)d_in[24];
  const float* eeg1_W   = (const float*)d_in[25];
  const float* eeg1_b   = (const float*)d_in[26];
  const float* eeg2_W   = (const float*)d_in[27];
  const float* eeg2_b   = (const float*)d_in[28];
  const int E = in_sizes[2];
  float* out = (float*)d_out;

  // ---- workspace carve-out ----
  char* ws = (char*)d_ws;
  size_t off = 0;
  auto take = [&](size_t bytes) { char* p = ws + off; off += al256(bytes); return p; };

  float*    XCUR  = (float*)take((size_t)MROWS * 64 * 4);
  float*    SKIP  = (float*)take((size_t)MROWS * 64 * 4);
  float*    RES   = (float*)take((size_t)MROWS * 64 * 4);   // also reused as E1
  _Float16* PF    = (_Float16*)take((size_t)BT * 64 * NP * 2);
  _Float16* PA    = (_Float16*)take((size_t)BT * 64 * NP * 2);
  _Float16* CIN   = (_Float16*)take((size_t)MROWS * 64 * 2); // also reused as E2 f16
  _Float16* GT    = (_Float16*)take((size_t)MROWS * 64 * 2);
  _Float16* ADJF  = (_Float16*)take((size_t)NP * NP * 2);
  _Float16* ADJA  = (_Float16*)take((size_t)NP * NP * 2);
  float*    ADJ32 = (float*)take((size_t)NNODE * NNODE * 4);
  float*    DINV  = (float*)take((size_t)NP * 4);
  _Float16* E3    = (_Float16*)take((size_t)BT * 256 * 2);
  _Float16* W_IN  = (_Float16*)take(4096 * 2);
  _Float16* WGF   = (_Float16*)take((size_t)NL * 4096 * 2);
  _Float16* WGA   = (_Float16*)take((size_t)NL * 4096 * 2);
  _Float16* WF0   = (_Float16*)take((size_t)NL * 4096 * 2);
  _Float16* WF1   = (_Float16*)take((size_t)NL * 4096 * 2);
  _Float16* WG0   = (_Float16*)take((size_t)NL * 4096 * 2);
  _Float16* WG1   = (_Float16*)take((size_t)NL * 4096 * 2);
  _Float16* WR    = (_Float16*)take((size_t)NL * 4096 * 2);
  _Float16* WS    = (_Float16*)take((size_t)NL * 4096 * 2);
  _Float16* WE1   = (_Float16*)take(4096 * 2);
  _Float16* WE2   = (_Float16*)take(4096 * 2);
  _Float16* WEEG1 = (_Float16*)take((size_t)256 * KEEG * 2);
  _Float16* WEEG2 = (_Float16*)take((size_t)64 * 256 * 2);
  (void)ws_size; (void)n_in; (void)out_size;

  const int small = (4096 + 255) / 256;
  // ---- weight prep (transpose/convert to f16 "Bt" layouts) ----
  k_cvt_T<<<small, 256, 0, stream>>>(W_IN, in_W, 64, 64);
  for (int i = 0; i < NL; ++i) {
    k_cvt_T<<<small, 256, 0, stream>>>(WGF + i * 4096, gcnf_W + i * 4096, 64, 64);
    k_cvt_T<<<small, 256, 0, stream>>>(WGA + i * 4096, gcna_W + i * 4096, 64, 64);
    k_cvt_S<<<small, 256, 0, stream>>>(WF0 + i * 4096, filt_W + i * 8192, 4096, 2, 0);
    k_cvt_S<<<small, 256, 0, stream>>>(WF1 + i * 4096, filt_W + i * 8192, 4096, 2, 1);
    k_cvt_S<<<small, 256, 0, stream>>>(WG0 + i * 4096, gate_W + i * 8192, 4096, 2, 0);
    k_cvt_S<<<small, 256, 0, stream>>>(WG1 + i * 4096, gate_W + i * 8192, 4096, 2, 1);
    k_cvt_S<<<small, 256, 0, stream>>>(WR + i * 4096, res_W + i * 4096, 4096, 1, 0);
    k_cvt_S<<<small, 256, 0, stream>>>(WS + i * 4096, skip_W + i * 4096, 4096, 1, 0);
  }
  k_cvt_T<<<small, 256, 0, stream>>>(WE1, end1_W, 64, 64);
  k_cvt_T<<<small, 256, 0, stream>>>(WE2, end2_W, 64, 64);
  k_cvt_T<<<(int)(((long)256 * KEEG + 255) / 256), 256, 0, stream>>>(WEEG1, eeg1_W, KEEG, 256);
  k_cvt_T<<<(256 * 64 + 255) / 256, 256, 0, stream>>>(WEEG2, eeg2_W, 256, 64);

  // ---- fixed adjacency ----
  k_fill_f32<<<(NNODE * NNODE + 255) / 256, 256, 0, stream>>>(ADJ32, (size_t)NNODE * NNODE);
  k_scatter<<<(E + 255) / 256, 256, 0, stream>>>(eidx, ew, E, ADJ32);
  k_rowdeg<<<NNODE, 32, 0, stream>>>(ADJ32, DINV);
  k_scale_cvt<<<(NP * NP + 255) / 256, 256, 0, stream>>>(ADJ32, DINV, ADJF);

  // ---- input projection + zero skip accumulator ----
  k_gemm64<<<MROWS / 64, 128, 0, stream>>>(x, W_IN, in_b, XCUR, (_Float16*)nullptr,
                                           (long)MROWS, 0);
  k_fill_f32<<<(int)(((size_t)MROWS * 64 + 255) / 256), 256, 0, stream>>>(SKIP, (size_t)MROWS * 64);

  // ---- layers ----
  for (int i = 0; i < NL; ++i) {
    int dil = 1 << (i % 4);
    // adaptive adjacency
    k_adapt<<<NNODE, NP, 0, stream>>>(emb_src + (size_t)i * NNODE * 16,
                                      emb_tgt + (size_t)i * NNODE * 16, ADJ32);
    k_rowdeg<<<NNODE, 32, 0, stream>>>(ADJ32, DINV);
    k_scale_cvt<<<(NP * NP + 255) / 256, 256, 0, stream>>>(ADJ32, DINV, ADJA);
    // GCN pre-projections (transposed f16 per slice)
    k_gcn_pre<<<dim3(BT, NP / 64), 128, 0, stream>>>(XCUR, WGF + i * 4096, PF);
    k_gcn_pre<<<dim3(BT, NP / 64), 128, 0, stream>>>(XCUR, WGA + i * 4096, PA);
    // adjacency GEMMs (fixed + adaptive fused)
    k_adj_gemm<<<dim3(BT, NP / 64), 128, 0, stream>>>(ADJF, ADJA, PF, PA,
                                                      gcnf_b + i * 64, gcna_b + i * 64, CIN);
    // gated dilated causal conv (4 waves cover the 4 t-tiles of one bn)
    k_conv_gemm<<<BN, 128, 0, stream>>>(CIN, WF1 + i * 4096, WF0 + i * 4096,
                                        WG1 + i * 4096, WG0 + i * 4096,
                                        filt_b + i * 64, gate_b + i * 64, dil, GT);
    // skip/res
    k_skipres<<<BN, 128, 0, stream>>>(GT, WS + i * 4096, WR + i * 4096,
                                      skip_b + i * 64, res_b + i * 64, SKIP, RES);
    // residual + layernorm
    k_layernorm<<<(MROWS + 7) / 8, 256, 0, stream>>>(XCUR, RES, ln_g + i * 64, ln_b + i * 64,
                                                     (long)MROWS);
  }

  // ---- output head ----
  // E1 = gelu(skip_sum) @ end1 + b   (reuse RES)
  k_gemm64<<<MROWS / 64, 128, 0, stream>>>(SKIP, WE1, end1_b, RES, (_Float16*)nullptr,
                                           (long)MROWS, 1);
  // E2 = gelu(E1) @ end2 + b, store f16 (reuse CIN; memory layout == (512 x 23680))
  k_gemm64<<<MROWS / 64, 128, 0, stream>>>(RES, WE2, end2_b, (float*)nullptr, CIN,
                                           (long)MROWS, 1);
  // eeg head (stage 1: 32 row-tiles x 4 waves covering 256 cols)
  k_eeg1<<<BT / 16, 128, 0, stream>>>(CIN, WEEG1, eeg1_b, E3);
  k_eeg2<<<BT / 64, 128, 0, stream>>>(E3, WEEG2, eeg2_b, out);
}
